// DGCNN_57054345560664
// MI455X (gfx1250) — compile-verified, matrix operand order checked by reading, not run on hardware
//
#include <hip/hip_runtime.h>
#include <hip/hip_bf16.h>

// ---------------------------------------------------------------------------
// DGCNN forward for MI455X (gfx1250, wave32, WMMA).
//   B=16, N=2048, K=20.  Edge-conv GEMMs run on v_wmma_f32_16x16x32_f16 with
//   f32 accumulation; BN is fused via a two-pass (stats, then apply) scheme so
//   the huge (B,N,K,O) tensor is never materialized.
//   Edge tiles are staged into LDS as packed v8h chunks (ds_store_b128) to
//   keep the DS pipe well below the matrix pipe.
// ---------------------------------------------------------------------------

typedef __attribute__((ext_vector_type(16))) _Float16 v16h;
typedef __attribute__((ext_vector_type(8)))  _Float16 v8h;
typedef __attribute__((ext_vector_type(8)))  float    v8f;

#define Bsz 16
#define Npts 2048
#define Kn 20
#define BN_EPS 1e-5f

// ---------------------------------------------------------------------------
// 0. transpose (B,3,N) -> (B,N,3)
// ---------------------------------------------------------------------------
__global__ __launch_bounds__(256) void transpose_in(const float* __restrict__ data,
                                                    float* __restrict__ x0) {
  int i = blockIdx.x * 256 + threadIdx.x;           // index into x0
  if (i < Bsz * Npts * 3) {
    int c = i % 3;
    int n = (i / 3) % Npts;
    int b = i / (3 * Npts);
    x0[i] = data[(b * 3 + c) * Npts + n];
  }
}

// ---------------------------------------------------------------------------
// 1. squared norms of feature rows
// ---------------------------------------------------------------------------
template <int C>
__global__ __launch_bounds__(256) void sqnorm_kernel(const float* __restrict__ X,
                                                     float* __restrict__ sq) {
  int i = blockIdx.x * 256 + threadIdx.x;
  if (i < Bsz * Npts) {
    const float* r = X + (size_t)i * C;
    float s = 0.f;
    for (int c = 0; c < C; ++c) s += r[c] * r[c];
    sq[i] = s;
  }
}

// ---------------------------------------------------------------------------
// 2. brute-force kNN (top-20 by neg squared distance), wave-per-row, fp32.
//    Distances for the wave's row live in LDS; 20 extract-max rounds with
//    wave32 shuffles (ties -> lower column index, matching lax.top_k).
// ---------------------------------------------------------------------------
template <int C>
__global__ __launch_bounds__(128) void knn_kernel(const float* __restrict__ X,
                                                  const float* __restrict__ sq,
                                                  int* __restrict__ idx) {
  __shared__ float xr[4][C];
  __shared__ float dist[4][64][32];                 // 32 KB
  const int w = threadIdx.x >> 5;
  const int lane = threadIdx.x & 31;
  const int r = blockIdx.x * 4 + w;                 // global row in [0, B*N)
  const int b = r >> 11;
  const float* xrow = X + (size_t)r * C;
  for (int c = lane; c < C; c += 32) xr[w][c] = xrow[c];
  __syncthreads();
  const float sqr = sq[r];
  const float* xb = X + (size_t)b * Npts * C;
  const float* sqb = sq + b * Npts;
  for (int j = 0; j < 64; ++j) {
    int col = j * 32 + lane;
    const float* xc = xb + (size_t)col * C;
    float dot = 0.f;
    if constexpr ((C & 3) == 0) {
      for (int c = 0; c < C; c += 4) {
        float4 v = *(const float4*)(xc + c);
        dot += v.x * xr[w][c] + v.y * xr[w][c + 1] + v.z * xr[w][c + 2] + v.w * xr[w][c + 3];
      }
    } else {
      for (int c = 0; c < C; ++c) dot += xc[c] * xr[w][c];
    }
    dist[w][j][lane] = 2.f * dot - sqr - sqb[col];
  }
  int* out = idx + (size_t)r * Kn;
  for (int t = 0; t < Kn; ++t) {
    float best = -INFINITY;
    int bj = 0;
    for (int j = 0; j < 64; ++j) {
      float v = dist[w][j][lane];
      if (v > best) { best = v; bj = j; }
    }
    int bcol = bj * 32 + lane;
    for (int m = 16; m > 0; m >>= 1) {
      float ov = __shfl_xor(best, m, 32);
      int   oc = __shfl_xor(bcol, m, 32);
      if (ov > best || (ov == best && oc < bcol)) { best = ov; bcol = oc; }
    }
    if (lane == 0) out[t] = bcol;
    if ((bcol & 31) == lane) dist[w][bcol >> 5][lane] = -INFINITY;
  }
}

// ---------------------------------------------------------------------------
// 3. weights fp32 -> padded f16 (Cout x Kp), Kp multiple of 32
// ---------------------------------------------------------------------------
__global__ __launch_bounds__(256) void wconvert(const float* __restrict__ W,
                                                _Float16* __restrict__ Wh,
                                                int Cout, int Kp, int C2) {
  int i = blockIdx.x * 256 + threadIdx.x;
  if (i < Cout * Kp) {
    int o = i / Kp, c = i % Kp;
    Wh[i] = (c < C2) ? (_Float16)W[o * C2 + c] : (_Float16)0.f;
  }
}

// ---------------------------------------------------------------------------
// 4. edge-conv GEMM on WMMA.
//    PASS==1 : accumulate per-channel sum / sumsq of h (for BN stats)
//    PASS==2 : h -> BN(scale,shift) -> leaky-relu -> max over k -> out
//    One workgroup = (batch b, 16 points). Edge tile (320 rows x Kp halfs)
//    staged in LDS once (packed b128 stores); each wave owns 16x16 tiles.
//    Fragment fine layout (per ISA 16-bit A table, B symmetric):
//      lane L, elem i<8 : K = kc + 8*(L>=16) + i ; elem i>=8 : +16
//    so each fragment = two contiguous 16B loads.
// ---------------------------------------------------------------------------
template <int Cin, int Cout, int Kp, int PASS>
__global__ __launch_bounds__(256) void edge_gemm(
    const float* __restrict__ X, const int* __restrict__ idx,
    const _Float16* __restrict__ Wh,
    const float* __restrict__ scale, const float* __restrict__ shift,
    float* __restrict__ out, float* __restrict__ gsum, float* __restrict__ gsumsq) {
  extern __shared__ char smem[];
  _Float16* eT = (_Float16*)smem;                                  // 320*Kp halfs
  float* aux = (float*)(smem + (size_t)320 * Kp * sizeof(_Float16));
  const int tid = threadIdx.x;
  const int b = blockIdx.x >> 7;                                   // N/16 = 128 tiles
  const int n0 = (blockIdx.x & 127) * 16;
  if (PASS == 1) {
    for (int i = tid; i < 2 * Cout; i += 256) aux[i] = 0.f;
  } else {
    unsigned* fm = (unsigned*)aux;
    for (int i = tid; i < 16 * Cout; i += 256) fm[i] = 0u;
  }
  // ---- stage edge features: rows = n_local*20 + k, cols = [nbr-ctr | ctr]
  //      packed 8 halfs per ds_store_b128.
  const size_t bN = (size_t)b * Npts;
  constexpr int CHUNKS = Kp / 8;                    // 8-half chunks per row
  for (int item = tid; item < 16 * Kn * CHUNKS; item += 256) {
    const int row = item / CHUNKS;
    const int ch = item % CHUNKS;
    const int nl = row / Kn, k = row % Kn;
    const int n = n0 + nl;
    const int nb = idx[(bN + n) * Kn + k];
    const float* ctr = X + (bN + n) * Cin;
    const float* nbr = X + (bN + nb) * Cin;
    const int c0 = ch * 8;
    v8h val;
#pragma unroll
    for (int j = 0; j < 8; ++j) {
      const int c = c0 + j;
      float v;
      if (c < Cin)            v = nbr[c] - ctr[c];
      else if (c < 2 * Cin)   v = ctr[c - Cin];
      else                    v = 0.f;
      val[j] = (_Float16)v;
    }
    *(v8h*)(eT + (size_t)row * Kp + c0) = val;
  }
  __syncthreads();
  const int wave = tid >> 5, lane = tid & 31;
  const int oL = lane & 15;
  const int hi = (lane & 16) ? 8 : 0;
  const int nTiles = Kn * (Cout / 16);
  for (int t = wave; t < nTiles; t += 8) {
    const int mt = t / (Cout / 16);
    const int ot = t % (Cout / 16);
    const int o = ot * 16 + oL;
    const _Float16* ea = eT + ((size_t)(mt * 16 + oL)) * Kp + hi;
    const _Float16* wb = Wh + (size_t)o * Kp + hi;
    v8f acc = {};
    for (int kc = 0; kc < Kp; kc += 32) {
      v8h a0 = *(const v8h*)(ea + kc);
      v8h a1 = *(const v8h*)(ea + kc + 16);
      v8h b0 = *(const v8h*)(wb + kc);
      v8h b1 = *(const v8h*)(wb + kc + 16);
      v16h a = __builtin_shufflevector(a0, a1, 0, 1, 2, 3, 4, 5, 6, 7, 8, 9, 10, 11, 12, 13, 14, 15);
      v16h bb = __builtin_shufflevector(b0, b1, 0, 1, 2, 3, 4, 5, 6, 7, 8, 9, 10, 11, 12, 13, 14, 15);
      acc = __builtin_amdgcn_wmma_f32_16x16x32_f16(false, a, false, bb, (short)0, acc, false, false);
    }
    if (PASS == 1) {
      float s = 0.f, q = 0.f;
#pragma unroll
      for (int r2 = 0; r2 < 8; ++r2) { float v = acc[r2]; s += v; q += v * v; }
      atomicAdd(&aux[o], s);
      atomicAdd(&aux[Cout + o], q);
    } else {
      unsigned* fm = (unsigned*)aux;
      const float sc = scale[o], sh = shift[o];
#pragma unroll
      for (int r2 = 0; r2 < 8; ++r2) {
        int row = mt * 16 + r2 + hi;
        int nl = row / Kn;
        float y = acc[r2] * sc + sh;
        y = (y >= 0.f) ? y : 0.2f * y;
        unsigned u = __float_as_uint(y);
        unsigned key = (u & 0x80000000u) ? ~u : (u | 0x80000000u);
        atomicMax(&fm[nl * Cout + o], key);
      }
    }
  }
  __syncthreads();
  if (PASS == 1) {
    for (int i = tid; i < Cout; i += 256) {
      atomicAdd(&gsum[i], aux[i]);
      atomicAdd(&gsumsq[i], aux[Cout + i]);
    }
  } else {
    unsigned* fm = (unsigned*)aux;
    for (int i = tid; i < 16 * Cout; i += 256) {
      int nl = i / Cout, o = i % Cout;
      unsigned key = fm[i];
      unsigned u = (key & 0x80000000u) ? (key & 0x7fffffffu) : ~key;
      out[(bN + n0 + nl) * Cout + o] = __uint_as_float(u);
    }
  }
}

// ---------------------------------------------------------------------------
// 5. BN stats -> (scale, shift)
// ---------------------------------------------------------------------------
__global__ __launch_bounds__(64) void finalize_bn(const float* __restrict__ gsum,
                                                  const float* __restrict__ gsumsq,
                                                  const float* __restrict__ gamma,
                                                  const float* __restrict__ beta,
                                                  float* __restrict__ scale,
                                                  float* __restrict__ shift,
                                                  int Cout, float invR) {
  int o = blockIdx.x * 64 + threadIdx.x;
  if (o < Cout) {
    float m = gsum[o] * invR;
    float v = gsumsq[o] * invR - m * m;
    float rs = rsqrtf(v + BN_EPS);
    float sc = gamma[o] * rs;
    scale[o] = sc;
    shift[o] = beta[o] - m * sc;
  }
}

// ---------------------------------------------------------------------------
// 6. pointwise conv (cat(512) -> 1024), two-pass BN, then max/mean pooling.
// ---------------------------------------------------------------------------
__device__ __forceinline__ float cat_dot(const float* __restrict__ d1,
                                         const float* __restrict__ d2,
                                         const float* __restrict__ d3,
                                         const float* __restrict__ d4,
                                         size_t bn, const float* __restrict__ w) {
  float h = 0.f;
  const float* r1 = d1 + bn * 64;
  for (int c = 0; c < 64; c += 4) {
    float4 v = *(const float4*)(r1 + c);
    h += v.x * w[c] + v.y * w[c + 1] + v.z * w[c + 2] + v.w * w[c + 3];
  }
  const float* r2 = d2 + bn * 64;
  for (int c = 0; c < 64; c += 4) {
    float4 v = *(const float4*)(r2 + c);
    h += v.x * w[64 + c] + v.y * w[64 + c + 1] + v.z * w[64 + c + 2] + v.w * w[64 + c + 3];
  }
  const float* r3 = d3 + bn * 128;
  for (int c = 0; c < 128; c += 4) {
    float4 v = *(const float4*)(r3 + c);
    h += v.x * w[128 + c] + v.y * w[128 + c + 1] + v.z * w[128 + c + 2] + v.w * w[128 + c + 3];
  }
  const float* r4 = d4 + bn * 256;
  for (int c = 0; c < 256; c += 4) {
    float4 v = *(const float4*)(r4 + c);
    h += v.x * w[256 + c] + v.y * w[256 + c + 1] + v.z * w[256 + c + 2] + v.w * w[256 + c + 3];
  }
  return h;
}

__global__ __launch_bounds__(256) void conv5_pass1(const float* __restrict__ d1,
                                                   const float* __restrict__ d2,
                                                   const float* __restrict__ d3,
                                                   const float* __restrict__ d4,
                                                   const float* __restrict__ W5,
                                                   float* __restrict__ gsum,
                                                   float* __restrict__ gsumsq) {
  const int tid = threadIdx.x;
  const int o = blockIdx.x * 64 + (tid & 63);
  const int b = blockIdx.y;
  const int ns = tid >> 6;
  const float* w = W5 + (size_t)o * 512;
  float s = 0.f, q = 0.f;
  for (int n = ns; n < Npts; n += 4) {
    float h = cat_dot(d1, d2, d3, d4, (size_t)b * Npts + n, w);
    s += h; q += h * h;
  }
  __shared__ float rs[256], rq[256];
  rs[tid] = s; rq[tid] = q;
  __syncthreads();
  if (tid < 64) {
    s = rs[tid] + rs[tid + 64] + rs[tid + 128] + rs[tid + 192];
    q = rq[tid] + rq[tid + 64] + rq[tid + 128] + rq[tid + 192];
    atomicAdd(&gsum[o], s);
    atomicAdd(&gsumsq[o], q);
  }
}

__global__ __launch_bounds__(256) void conv5_pass2(const float* __restrict__ d1,
                                                   const float* __restrict__ d2,
                                                   const float* __restrict__ d3,
                                                   const float* __restrict__ d4,
                                                   const float* __restrict__ W5,
                                                   const float* __restrict__ scale,
                                                   const float* __restrict__ shift,
                                                   float* __restrict__ fbuf) {
  const int tid = threadIdx.x;
  const int o = blockIdx.x * 64 + (tid & 63);
  const int b = blockIdx.y;
  const int ns = tid >> 6;
  const float* w = W5 + (size_t)o * 512;
  const float sc = scale[o], sh = shift[o];
  float mx = -INFINITY, sm = 0.f;
  for (int n = ns; n < Npts; n += 4) {
    float h = cat_dot(d1, d2, d3, d4, (size_t)b * Npts + n, w);
    float y = h * sc + sh;
    y = (y >= 0.f) ? y : 0.2f * y;
    mx = fmaxf(mx, y);
    sm += y;
  }
  __shared__ float rmx[256], rsm[256];
  rmx[tid] = mx; rsm[tid] = sm;
  __syncthreads();
  if (tid < 64) {
    mx = fmaxf(fmaxf(rmx[tid], rmx[tid + 64]), fmaxf(rmx[tid + 128], rmx[tid + 192]));
    sm = rsm[tid] + rsm[tid + 64] + rsm[tid + 128] + rsm[tid + 192];
    fbuf[(size_t)b * 2048 + o] = mx;
    fbuf[(size_t)b * 2048 + 1024 + o] = sm * (1.f / (float)Npts);
  }
}

// ---------------------------------------------------------------------------
// 7. MLP head (16 rows) — single workgroup, LDS resident.
// ---------------------------------------------------------------------------
__global__ __launch_bounds__(512) void mlp_head(const float* __restrict__ fbuf,
                                                const float* __restrict__ fc1W,
                                                const float* __restrict__ bn1g,
                                                const float* __restrict__ bn1b,
                                                const float* __restrict__ fc2W,
                                                const float* __restrict__ fc2b,
                                                const float* __restrict__ bn2g,
                                                const float* __restrict__ bn2b,
                                                const float* __restrict__ fc3W,
                                                const float* __restrict__ fc3b,
                                                float* __restrict__ out) {
  __shared__ float h1[16][512];
  __shared__ float h2[16][256];
  const int tid = threadIdx.x;
  {
    int j = tid;                                    // 512 columns
    const float* w = fc1W + (size_t)j * 2048;
    float v[16];
#pragma unroll
    for (int r = 0; r < 16; ++r) {
      const float* fr = fbuf + (size_t)r * 2048;
      float s = 0.f;
      for (int c = 0; c < 2048; c += 4) {
        float4 a = *(const float4*)(fr + c);
        float4 ww = *(const float4*)(w + c);
        s += a.x * ww.x + a.y * ww.y + a.z * ww.z + a.w * ww.w;
      }
      v[r] = s;
    }
    float m = 0.f, q = 0.f;
#pragma unroll
    for (int r = 0; r < 16; ++r) { m += v[r]; q += v[r] * v[r]; }
    m *= (1.f / 16.f); q = q * (1.f / 16.f) - m * m;
    float rs = rsqrtf(q + BN_EPS);
    float g = bn1g[j], bb = bn1b[j];
#pragma unroll
    for (int r = 0; r < 16; ++r) {
      float y = (v[r] - m) * rs * g + bb;
      h1[r][j] = (y >= 0.f) ? y : 0.2f * y;
    }
  }
  __syncthreads();
  if (tid < 256) {
    int j = tid;
    const float* w = fc2W + (size_t)j * 512;
    float v[16];
#pragma unroll
    for (int r = 0; r < 16; ++r) {
      float s = 0.f;
      for (int c = 0; c < 512; ++c) s += h1[r][c] * w[c];
      v[r] = s + fc2b[j];
    }
    float m = 0.f, q = 0.f;
#pragma unroll
    for (int r = 0; r < 16; ++r) { m += v[r]; q += v[r] * v[r]; }
    m *= (1.f / 16.f); q = q * (1.f / 16.f) - m * m;
    float rs = rsqrtf(q + BN_EPS);
    float g = bn2g[j], bb = bn2b[j];
#pragma unroll
    for (int r = 0; r < 16; ++r) {
      float y = (v[r] - m) * rs * g + bb;
      h2[r][j] = (y >= 0.f) ? y : 0.2f * y;
    }
  }
  __syncthreads();
  if (tid < 40) {
    int j = tid;
    const float* w = fc3W + (size_t)j * 256;
    for (int r = 0; r < 16; ++r) {
      float s = 0.f;
      for (int c = 0; c < 256; ++c) s += h2[r][c] * w[c];
      out[r * 40 + j] = s + fc3b[j];
    }
  }
}

// ---------------------------------------------------------------------------
// Host orchestration
// ---------------------------------------------------------------------------
template <int Cin, int Cout, int Kp>
static void run_edge_conv(const float* X, const float* W, const float* g, const float* bta,
                          float* outFeat, float* sqb, int* idxb, _Float16* whb,
                          float* stats, float* scsh, hipStream_t s) {
  sqnorm_kernel<Cin><<<(Bsz * Npts + 255) / 256, 256, 0, s>>>(X, sqb);
  knn_kernel<Cin><<<Bsz * Npts / 4, 128, 0, s>>>(X, sqb, idxb);
  wconvert<<<(Cout * Kp + 255) / 256, 256, 0, s>>>(W, whb, Cout, Kp, 2 * Cin);
  hipMemsetAsync(stats, 0, 2048 * sizeof(float), s);
  size_t lds1 = (size_t)320 * Kp * sizeof(_Float16) + (size_t)2 * Cout * sizeof(float);
  edge_gemm<Cin, Cout, Kp, 1><<<Bsz * (Npts / 16), 256, lds1, s>>>(
      X, idxb, whb, nullptr, nullptr, nullptr, stats, stats + 1024);
  finalize_bn<<<(Cout + 63) / 64, 64, 0, s>>>(stats, stats + 1024, g, bta,
                                              scsh, scsh + 1024, Cout,
                                              1.f / ((float)Bsz * Npts * Kn));
  size_t lds2 = (size_t)320 * Kp * sizeof(_Float16) + (size_t)16 * Cout * sizeof(unsigned);
  edge_gemm<Cin, Cout, Kp, 2><<<Bsz * (Npts / 16), 256, lds2, s>>>(
      X, idxb, whb, scsh, scsh + 1024, outFeat, nullptr, nullptr);
}

extern "C" void kernel_launch(void* const* d_in, const int* in_sizes, int n_in,
                              void* d_out, int out_size, void* d_ws, size_t ws_size,
                              hipStream_t stream) {
  const float* data = (const float*)d_in[0];
  const float* W1 = (const float*)d_in[1];  const float* g1 = (const float*)d_in[2];  const float* b1 = (const float*)d_in[3];
  const float* W2 = (const float*)d_in[4];  const float* g2 = (const float*)d_in[5];  const float* b2 = (const float*)d_in[6];
  const float* W3 = (const float*)d_in[7];  const float* g3 = (const float*)d_in[8];  const float* b3 = (const float*)d_in[9];
  const float* W4 = (const float*)d_in[10]; const float* g4 = (const float*)d_in[11]; const float* b4 = (const float*)d_in[12];
  const float* W5 = (const float*)d_in[13]; const float* g5 = (const float*)d_in[14]; const float* b5 = (const float*)d_in[15];
  const float* fc1W = (const float*)d_in[16];
  const float* bn1g = (const float*)d_in[17]; const float* bn1b = (const float*)d_in[18];
  const float* fc2W = (const float*)d_in[19]; const float* fc2b = (const float*)d_in[20];
  const float* bn2g = (const float*)d_in[21]; const float* bn2b = (const float*)d_in[22];
  const float* fc3W = (const float*)d_in[23]; const float* fc3b = (const float*)d_in[24];
  float* out = (float*)d_out;

  // workspace carve-up (all offsets 256B aligned)
  char* ws = (char*)d_ws;
  size_t off = 0;
  auto alloc = [&](size_t bytes) {
    char* p = ws + off;
    off = (off + bytes + 255) & ~(size_t)255;
    return p;
  };
  float* x0   = (float*)alloc((size_t)Bsz * Npts * 3 * 4);
  float* d1   = (float*)alloc((size_t)Bsz * Npts * 64 * 4);
  float* d2   = (float*)alloc((size_t)Bsz * Npts * 64 * 4);
  float* d3   = (float*)alloc((size_t)Bsz * Npts * 128 * 4);
  float* d4   = (float*)alloc((size_t)Bsz * Npts * 256 * 4);
  int*   idxb = (int*)alloc((size_t)Bsz * Npts * Kn * 4);
  float* sqb  = (float*)alloc((size_t)Bsz * Npts * 4);
  _Float16* whb = (_Float16*)alloc((size_t)256 * 256 * 2);
  float* stats = (float*)alloc((size_t)2048 * 4);   // [sum(1024) | sumsq(1024)]
  float* scsh  = (float*)alloc((size_t)2048 * 4);   // [scale(1024) | shift(1024)]
  float* fbuf  = (float*)alloc((size_t)Bsz * 2048 * 4);
  (void)ws_size; (void)in_sizes; (void)n_in; (void)out_size;

  transpose_in<<<(Bsz * Npts * 3 + 255) / 256, 256, 0, stream>>>(data, x0);

  run_edge_conv<3,   64,  32 >(x0, W1, g1, b1, d1, sqb, idxb, whb, stats, scsh, stream);
  run_edge_conv<64,  64,  128>(d1, W2, g2, b2, d2, sqb, idxb, whb, stats, scsh, stream);
  run_edge_conv<64,  128, 128>(d2, W3, g3, b3, d3, sqb, idxb, whb, stats, scsh, stream);
  run_edge_conv<128, 256, 256>(d3, W4, g4, b4, d4, sqb, idxb, whb, stats, scsh, stream);

  // pointwise conv 512 -> 1024 with fused BN (two-pass) + max/mean pooling
  hipMemsetAsync(stats, 0, 2048 * sizeof(float), stream);
  dim3 g5grid(16, Bsz);
  conv5_pass1<<<g5grid, 256, 0, stream>>>(d1, d2, d3, d4, W5, stats, stats + 1024);
  finalize_bn<<<(1024 + 63) / 64, 64, 0, stream>>>(stats, stats + 1024, g5, b5,
                                                   scsh, scsh + 1024, 1024,
                                                   1.f / ((float)Bsz * Npts));
  conv5_pass2<<<g5grid, 256, 0, stream>>>(d1, d2, d3, d4, W5, scsh, scsh + 1024, fbuf);

  mlp_head<<<1, 512, 0, stream>>>(fbuf, fc1W, bn1g, bn1b, fc2W, fc2b, bn2g, bn2b,
                                  fc3W, fc3b, out);
}
